// ContentBasedAttention_8641474199608
// MI455X (gfx1250) — matrix operand order repaired and checked by reading, not compiled, not run on hardware
//
#include <hip/hip_runtime.h>
#include <hip/hip_bf16.h>
#include <math.h>

#define BB 32
#define SS 2048
#define DD 1024

#define BLK_M  64
#define APITCH 1048   // bf16 elems per A row in LDS (2096B rows, 16B aligned, conflict-free b128)
#define KCHUNK 64     // K elements staged per TDM chunk
#define BPITCH 88     // bf16 per staged B row: 64 data + 24 pad (TDM pad: 32DW interval, 12DW amount)
#define NTPC   64     // columns handled per outer iteration (4 x 16-wide tiles)

typedef __attribute__((ext_vector_type(16))) __bf16 v16bf;
typedef __attribute__((ext_vector_type(8)))  float  v8f;
typedef unsigned int u32x4 __attribute__((ext_vector_type(4)));
typedef int          i32x8 __attribute__((ext_vector_type(8)));
typedef int          i32x4 __attribute__((ext_vector_type(4)));

union Frag16 { unsigned int u[8]; v16bf v; };

__device__ __forceinline__ unsigned short f2bf(float f) {
  unsigned int u = __float_as_uint(f);
  u += 0x7FFFu + ((u >> 16) & 1u);          // round-to-nearest-even
  return (unsigned short)(u >> 16);
}

// TDM: DMA a 64(n-rows) x 64(K) bf16 tile of Wt (row stride 1024 elems) into LDS,
// padding each 128B row with 48B -> 176B pitch (= BPITCH bf16). ISA D# per 8.3/8.4.
// This toolchain exposes the 6-arg builtin: (g0, g1, g2, g3, g4, cpol).
__device__ __forceinline__ void tdm_issue_b_tile(unsigned long long gaddr, unsigned lds_off) {
  u32x4 g0 = { 1u,                                    // count = 1 valid descriptor
               lds_off,                               // lds_addr (bytes)
               (unsigned)gaddr,                       // global_addr[31:0]
               (unsigned)(gaddr >> 32) | (2u << 30) };// global_addr[56:32] | type=2
  i32x8 g1 = { (int)((1u << 16)      // data_size = 1 (2 bytes)
                   | (1u << 20)      // pad_enable
                   | (4u << 22)      // pad_interval: 32 DWORDs (128B row)
                   | (11u << 25)),   // pad_amount:   12 DWORDs (48B)
               (int)(64u << 16),     // tensor_dim0 = 64 (bits 79:48 low part)
               (int)(64u << 16),     // tensor_dim0 hi=0 | tensor_dim1 = 64 (low16)
               (int)(64u << 16),     // tensor_dim1 hi=0 | tile_dim0 = 64
               64,                   // tile_dim1 = 64 | tile_dim2 = 0
               1024,                 // tensor_dim0_stride low32 = 1024
               0, 0 };               // stride hi, tensor_dim1_stride
  i32x4 z4 = { 0, 0, 0, 0 };
  i32x8 z8 = { 0, 0, 0, 0, 0, 0, 0, 0 };
  __builtin_amdgcn_tensor_load_to_lds(g0, g1, z4, z4, z8, 0);
}

// ---------------------------------------------------------------------------
// W_enc [k][n] f32 -> Wt [n][k] bf16 (tiled transpose through LDS)
// ---------------------------------------------------------------------------
__global__ void wenc_transpose_kernel(const float* __restrict__ W,
                                      unsigned short* __restrict__ Wt) {
  __shared__ unsigned short tile[32][33];
  const int k0 = blockIdx.x * 32, n0 = blockIdx.y * 32;
  for (int i = threadIdx.y; i < 32; i += 8) {
    float v = W[(size_t)(k0 + i) * DD + n0 + threadIdx.x];
    tile[threadIdx.x][i] = f2bf(v);
  }
  __syncthreads();
  for (int i = threadIdx.y; i < 32; i += 8)
    Wt[(size_t)(n0 + i) * DD + k0 + threadIdx.x] = tile[i][threadIdx.x];
}

// ---------------------------------------------------------------------------
// Y[b][a] = X[b,:] . W[:,a] + bias[a] (+ bias2[a])   (32 x 1024 outputs)
// ---------------------------------------------------------------------------
__global__ void small_gemm_kernel(const float* __restrict__ X,
                                  const float* __restrict__ W,
                                  const float* __restrict__ bias,
                                  const float* __restrict__ bias2,
                                  float* __restrict__ Y) {
  const int a = blockIdx.x * 256 + threadIdx.x;
  const int b = blockIdx.y;
  float acc = bias[a] + (bias2 ? bias2[a] : 0.0f);
  const float* x = X + (size_t)b * DD;
  const float* w = W + a;
#pragma unroll 4
  for (int e = 0; e < DD; ++e) acc += x[e] * w[(size_t)e * DD];
  Y[(size_t)b * DD + a] = acc;
}

// ---------------------------------------------------------------------------
// Fused: scores[b,s] = sum_a w_attn[a] * tanh( (enc @ W_enc)[b,s,a] + decb[b,a] )
// 1024 WGs x 256 threads (8 waves, 2/SIMD). A tile (64x1024) bf16 resident in
// LDS; B double-buffered via TDM tensor_load_to_lds, pipelined one chunk ahead.
// Wave w: row group (w>>1), column half (w&1); 2 WMMAs per A fragment.
// ---------------------------------------------------------------------------
__global__ __launch_bounds__(256, 1)
void attn_scores_kernel(const float* __restrict__ enc,
                        const unsigned short* __restrict__ Wt,
                        const float* __restrict__ decb,
                        const float* __restrict__ w_attn,
                        float* __restrict__ scores) {
  __shared__ __align__(16) unsigned short sA[BLK_M * APITCH];   // 134,144 B
  __shared__ __align__(16) unsigned short sB[2][NTPC * BPITCH]; // 2 x 11,264 B
  __shared__ float sPart[2][BLK_M];                             // 512 B

  const int tid   = threadIdx.x;
  const int lane  = tid & 31;
  const int wave  = tid >> 5;
  const int g     = wave >> 1;                 // row group (0..3)
  const int h     = wave & 1;                  // column half (0..1)
  const size_t rowBase = (size_t)blockIdx.x * BLK_M;
  const int b = (int)(rowBase >> 11);          // S = 2048; tiles never straddle b

  const unsigned long long wbase = (unsigned long long)(const void*)Wt;

  // prologue: kick off the first B tile DMA before touching A
  if (wave == 0) tdm_issue_b_tile(wbase, (unsigned)(size_t)&sB[0][0]);

  // ---- load + convert A tile: 64 rows x 1024 f32 -> bf16 LDS ----
  for (int i = tid; i < BLK_M * (DD / 4); i += 256) {
    const int r  = i >> 8;
    const int c4 = i & 255;
    const float4 f = ((const float4*)(enc + (rowBase + r) * DD))[c4];
    uint2 p;
    p.x = (unsigned)f2bf(f.x) | ((unsigned)f2bf(f.y) << 16);
    p.y = (unsigned)f2bf(f.z) | ((unsigned)f2bf(f.w) << 16);
    *(uint2*)&sA[r * APITCH + c4 * 4] = p;
  }

  const int mRow  = g * 16 + (lane & 15);
  const int khalf = lane >> 4;
  const int ncol  = lane & 15;

  float accS[8];
#pragma unroll
  for (int r = 0; r < 8; ++r) accS[r] = 0.0f;

  for (int ntp = 0; ntp < DD / NTPC; ++ntp) {    // 16 outer column steps
    const int n0 = ntp * NTPC;
    v8f C0 = {0,0,0,0,0,0,0,0};
    v8f C1 = {0,0,0,0,0,0,0,0};

    for (int c = 0; c < DD / KCHUNK; ++c) {      // 16 K-chunks of 64
      if (wave == 0) __builtin_amdgcn_s_wait_tensorcnt(0);  // chunk c landed
      __syncthreads();   // data visible; everyone done with buffer (c+1)&1
      if (wave == 0) {   // pipeline: DMA next chunk while we compute this one
        int nc = c + 1, nn = n0;
        if (nc == DD / KCHUNK) { nc = 0; nn = n0 + NTPC; }
        if (nn < DD)
          tdm_issue_b_tile(wbase + 2ull * ((unsigned long long)nn * DD + (unsigned)(nc * KCHUNK)),
                           (unsigned)(size_t)&sB[(c + 1) & 1][0]);
      }
      const unsigned short* bbase = &sB[c & 1][0];
#pragma unroll
      for (int kk = 0; kk < KCHUNK / 32; ++kk) { // 2 WMMA K-steps
        const int kb = c * KCHUNK + kk * 32;
        Frag16 afr, b0, b1;
        const unsigned short* ap = &sA[mRow * APITCH + kb + 16 * khalf];
        *(uint4*)&afr.u[0] = *(const uint4*)(ap);
        *(uint4*)&afr.u[4] = *(const uint4*)(ap + 8);
        const int kloc = kk * 32 + 16 * khalf;
        const unsigned short* bp0 = bbase + (h * 32 + ncol) * BPITCH + kloc;
        const unsigned short* bp1 = bbase + (h * 32 + 16 + ncol) * BPITCH + kloc;
        *(uint4*)&b0.u[0] = *(const uint4*)(bp0);
        *(uint4*)&b0.u[4] = *(const uint4*)(bp0 + 8);
        *(uint4*)&b1.u[0] = *(const uint4*)(bp1);
        *(uint4*)&b1.u[4] = *(const uint4*)(bp1 + 8);
        C0 = __builtin_amdgcn_wmma_f32_16x16x32_bf16(false, afr.v, false, b0.v,
                                                     (short)0, C0, false, false);
        C1 = __builtin_amdgcn_wmma_f32_16x16x32_bf16(false, afr.v, false, b1.v,
                                                     (short)0, C1, false, false);
      }
    }

    // epilogue: tanh + weighted column reduction (C: VGPR r -> row r+8*khalf)
    const int na = n0 + h * 32 + ncol;
    const int nb = na + 16;
    const float da = decb[(size_t)b * DD + na];
    const float db = decb[(size_t)b * DD + nb];
    const float wa = w_attn[na];
    const float wb = w_attn[nb];
#pragma unroll
    for (int r = 0; r < 8; ++r) {
      accS[r] += wa * tanhf(C0[r] + da);
      accS[r] += wb * tanhf(C1[r] + db);
    }
  }

  // column sum across the 16 lanes of each half
#pragma unroll
  for (int r = 0; r < 8; ++r) {
    float v = accS[r];
    v += __shfl_xor(v, 1, 32);
    v += __shfl_xor(v, 2, 32);
    v += __shfl_xor(v, 4, 32);
    v += __shfl_xor(v, 8, 32);
    accS[r] = v;
  }
  if (ncol == 0) {
    const int base = g * 16 + 8 * khalf;
#pragma unroll
    for (int r = 0; r < 8; ++r) sPart[h][base + r] = accS[r];
  }
  __syncthreads();
  if (tid < BLK_M)
    scores[rowBase + tid] = sPart[0][tid] + sPart[1][tid];
}

// ---------------------------------------------------------------------------
// Masked softmax over S per batch row
// ---------------------------------------------------------------------------
__global__ void softmax_kernel(const float* __restrict__ scores,
                               const int* __restrict__ enc_len,
                               float* __restrict__ attn) {
  const int b = blockIdx.x;
  const int tid = threadIdx.x;            // 256 threads = 8 waves
  __shared__ float red[8];
  __shared__ float bc[2];
  const int len = enc_len[b];
  const float* sr = scores + (size_t)b * SS;

  float mx = -3.402823466e38f;
  for (int s = tid; s < SS; s += 256)
    if (s < len) mx = fmaxf(mx, sr[s]);
  for (int o = 16; o; o >>= 1) mx = fmaxf(mx, __shfl_xor(mx, o, 32));
  if ((tid & 31) == 0) red[tid >> 5] = mx;
  __syncthreads();
  if (tid == 0) {
    float m = red[0];
    for (int i = 1; i < 8; ++i) m = fmaxf(m, red[i]);
    bc[0] = m;
  }
  __syncthreads();
  mx = bc[0];

  float sum = 0.0f;
  for (int s = tid; s < SS; s += 256)
    if (s < len) sum += __expf(sr[s] - mx);
  for (int o = 16; o; o >>= 1) sum += __shfl_xor(sum, o, 32);
  if ((tid & 31) == 0) red[tid >> 5] = sum;
  __syncthreads();
  if (tid == 0) {
    float t = 0.0f;
    for (int i = 0; i < 8; ++i) t += red[i];
    bc[1] = 1.0f / t;
  }
  __syncthreads();
  const float inv = bc[1];
  for (int s = tid; s < SS; s += 256)
    attn[(size_t)b * SS + s] = (s < len) ? __expf(sr[s] - mx) * inv : 0.0f;
}

// ---------------------------------------------------------------------------
// context_raw[b][e] = sum_s attn[b,s] * enc[b,s,e]   (memory-bound stream)
// ---------------------------------------------------------------------------
__global__ void context_kernel(const float* __restrict__ enc,
                               const float* __restrict__ attn,
                               float* __restrict__ ctx) {
  __shared__ float sa[SS];
  const int b = blockIdx.x;
  const int e = blockIdx.y * 256 + threadIdx.x;
  for (int s = threadIdx.x; s < SS; s += 256) sa[s] = attn[(size_t)b * SS + s];
  __syncthreads();
  const float* ep = enc + ((size_t)b * SS) * DD + e;
  float acc = 0.0f;
#pragma unroll 8
  for (int s = 0; s < SS; ++s) acc += sa[s] * ep[(size_t)s * DD];
  ctx[(size_t)b * DD + e] = acc;
}

// ---------------------------------------------------------------------------
extern "C" void kernel_launch(void* const* d_in, const int* in_sizes, int n_in,
                              void* d_out, int out_size, void* d_ws, size_t ws_size,
                              hipStream_t stream) {
  const float* enc     = (const float*)d_in[0];   // [32,2048,1024]
  const float* dec     = (const float*)d_in[1];   // [32,1024]
  const float* W_enc   = (const float*)d_in[2];   // [1024,1024]
  const float* b_enc   = (const float*)d_in[3];   // [1024]
  const float* W_dec   = (const float*)d_in[4];   // [1024,1024]
  const float* b_dec   = (const float*)d_in[5];   // [1024]
  const float* w_att   = (const float*)d_in[6];   // [1024]
  const float* W_out   = (const float*)d_in[7];   // [1024,1024]
  const float* b_out   = (const float*)d_in[8];   // [1024]
  const int*   enc_len = (const int*)d_in[9];     // [32]

  float* ctx_out  = (float*)d_out;                // [32,1024] (first tuple output)
  float* attn_out = (float*)d_out + BB * DD;      // [32,2048] (second tuple output)

  char* ws = (char*)d_ws;
  unsigned short* Wt = (unsigned short*)ws;                          // 2 MB bf16
  float* decb        = (float*)(ws + 2 * 1024 * 1024);               // 128 KB
  float* scores      = (float*)(ws + 2 * 1024 * 1024 + 128 * 1024);  // 256 KB
  float* ctx_raw     = (float*)(ws + 2 * 1024 * 1024 + 384 * 1024);  // 128 KB

  wenc_transpose_kernel<<<dim3(DD / 32, DD / 32), dim3(32, 8), 0, stream>>>(W_enc, Wt);
  small_gemm_kernel<<<dim3(DD / 256, BB), 256, 0, stream>>>(dec, W_dec, b_dec, b_enc, decb);
  attn_scores_kernel<<<dim3((BB * SS) / BLK_M), 256, 0, stream>>>(enc, Wt, decb, w_att, scores);
  softmax_kernel<<<dim3(BB), 256, 0, stream>>>(scores, enc_len, attn_out);
  context_kernel<<<dim3(BB, DD / 256), 256, 0, stream>>>(enc, attn_out, ctx_raw);
  small_gemm_kernel<<<dim3(DD / 256, BB), 256, 0, stream>>>(ctx_raw, W_out, b_out, nullptr, ctx_out);
}
